// MandiFlowNet_32495722561894
// MI455X (gfx1250) — matrix operand order repaired and checked by reading, not compiled, not run on hardware
//
#include <hip/hip_runtime.h>
#include <hip/hip_bf16.h>

// MI455X (gfx1250) implementation.
// Workspace layout (needs ~118 MB):
//   deg/dinv (N f32), norm (E+N f32), h0 (N*64 f32), h1 (N*64 f32),
//   seq (N*7*64 f32), bf16 weights (Wg2^T, Wih0, Whh0, Wih1, Whh1).
// LSTM cell state aliases h0 (free after GCN phase).

#define N_NODES 50000
#define N_EDGES 400000
#define T_STEPS 7
#define FEATD 7
#define HD 64
#define LN_EPS 1e-5f

typedef __attribute__((ext_vector_type(16))) __bf16 bf16x16;
typedef __attribute__((ext_vector_type(8))) float f32x8;
typedef __attribute__((ext_vector_type(8))) unsigned int u32x8;

__device__ __forceinline__ unsigned short f2bf_bits(float f) {
  union { float f; unsigned u; } v; v.f = f;
  unsigned r = v.u + 0x7FFFu + ((v.u >> 16) & 1u);   // round-to-nearest-even
  return (unsigned short)(r >> 16);
}
__device__ __forceinline__ unsigned pack2bf(float lo, float hi) {
  return (unsigned)f2bf_bits(lo) | ((unsigned)f2bf_bits(hi) << 16);
}
__device__ __forceinline__ float leaky01(float x) { return x > 0.f ? x : 0.01f * x; }
__device__ __forceinline__ float sigm(float x) { return 1.f / (1.f + __expf(-x)); }

// ---- WMMA fragment loaders (per CDNA5 ISA 7.12.2 VGPR layouts) ----
// A 16x32 bf16 from an fp32 row (32 K-values starting at `row`):
//   lane 0-15: row M=lane, VGPR v<4 -> K=2v,2v+1 ; v>=4 -> K=2v+8,2v+9
//   lane 16-31: same M, K offset +8
template <bool LEAKY>
__device__ __forceinline__ bf16x16 load_a_frag(const float* row, int lane) {
  const int khalf = (lane & 16) ? 8 : 0;
  u32x8 au;
#pragma unroll
  for (int v = 0; v < 8; ++v) {
    int k = ((v < 4) ? (2 * v) : (8 + 2 * v)) + khalf;
    float x0 = row[k], x1 = row[k + 1];
    if (LEAKY) { x0 = leaky01(x0); x1 = leaky01(x1); }
    au[v] = pack2bf(x0, x1);
  }
  return __builtin_bit_cast(bf16x16, au);
}
// B 32x16 bf16 from column-major bf16 weights; `bt` = base + col*64 + kchunk.
//   lane 0-15: col N=lane, VGPR v<4 -> K=2v,2v+1 ; v>=4 -> K=2v+8,2v+9 ; lane 16-31: K offset +8
__device__ __forceinline__ bf16x16 load_b_frag(const unsigned short* bt, int lane) {
  const int khalf = (lane & 16) ? 8 : 0;
  u32x8 bu;
#pragma unroll
  for (int v = 0; v < 8; ++v)
    bu[v] = *(const unsigned*)(bt + khalf + ((v < 4) ? (2 * v) : (8 + 2 * v)));
  return __builtin_bit_cast(bf16x16, bu);
}

// ---------------- weight conversion ----------------
__global__ void k_f32_to_bf16(const float* __restrict__ src, unsigned short* __restrict__ dst, int n) {
  int i = blockIdx.x * 256 + threadIdx.x;
  if (i < n) dst[i] = f2bf_bits(src[i]);
}
// Wg2 is (K=64, Ncol=64) row-major; store B^T so column j is contiguous in K.
__global__ void k_wg2_transpose_bf16(const float* __restrict__ src, unsigned short* __restrict__ dst) {
  int i = blockIdx.x * 256 + threadIdx.x;  // 4096
  if (i < HD * HD) {
    int k = i >> 6, j = i & 63;
    dst[j * HD + k] = f2bf_bits(src[k * HD + j]);
  }
}

// ---------------- gcn_norm ----------------
__global__ void k_deg_init(float* deg) {
  int i = blockIdx.x * 256 + threadIdx.x;
  if (i < N_NODES) deg[i] = 1.0f;  // self-loop weight
}
__global__ void k_deg_edge(const int* __restrict__ ei, const float* __restrict__ w, float* deg) {
  int e = blockIdx.x * 256 + threadIdx.x;
  if (e < N_EDGES) atomicAdd(&deg[ei[N_EDGES + e]], w[e]);
}
__global__ void k_dinv_selfnorm(float* deg, float* norm) {
  int i = blockIdx.x * 256 + threadIdx.x;
  if (i < N_NODES) {
    float d = deg[i];
    float di = d > 0.f ? rsqrtf(d) : 0.f;
    deg[i] = di;                      // deg now holds dinv
    norm[N_EDGES + i] = di * di;      // self-loop norm
  }
}
__global__ void k_norm_edge(const int* __restrict__ ei, const float* __restrict__ w,
                            const float* __restrict__ dinv, float* __restrict__ norm) {
  int e = blockIdx.x * 256 + threadIdx.x;
  if (e < N_EDGES) norm[e] = dinv[ei[e]] * w[e] * dinv[ei[N_EDGES + e]];
}

// ---------------- GCN phase ----------------
__global__ void k_xw1(const float* __restrict__ x, const float* __restrict__ Wg1,
                      float* __restrict__ h0, int t) {
  int i = blockIdx.x * 256 + threadIdx.x;
  if (i >= N_NODES * HD) return;
  int n = i >> 6, j = i & 63;
  const float* xr = x + (size_t)n * (T_STEPS * FEATD) + t * FEATD;
  float s = 0.f;
#pragma unroll
  for (int k = 0; k < FEATD; ++k) s += xr[k] * Wg1[k * HD + j];
  h0[i] = s;
}
__global__ void k_init_bias(float* __restrict__ h, const float* __restrict__ b) {
  int i = blockIdx.x * 256 + threadIdx.x;
  if (i < N_NODES * HD) h[i] = b[i & 63];
}
// out[dst] += norm * in[src]; 4 edges x 64 features per block. Edges >= E are self loops.
__global__ void k_scatter(const int* __restrict__ ei, const float* __restrict__ norm,
                          const float* __restrict__ hin, float* __restrict__ hout) {
  int e = blockIdx.x * 4 + (threadIdx.x >> 6);
  if (e >= N_EDGES + N_NODES) return;
  int f = threadIdx.x & 63;
  int s, d;
  if (e < N_EDGES) { s = ei[e]; d = ei[N_EDGES + e]; }
  else             { s = d = e - N_EDGES; }
  float nr = norm[e];
  atomicAdd(&hout[(size_t)d * HD + f], nr * hin[(size_t)s * HD + f]);
}
// hout = leaky_relu(hin) @ Wg2  (WMMA bf16; 16 rows x 64 cols per 4-wave block)
__global__ void __launch_bounds__(128) k_gemm_gcn(const float* __restrict__ hin,
                                                  const unsigned short* __restrict__ wbf,
                                                  float* __restrict__ hout) {
  int lane = threadIdx.x & 31, wv = threadIdx.x >> 5;
  int m0 = blockIdx.x * 16;
  const float* arow = hin + (size_t)(m0 + (lane & 15)) * HD;
  int colg = wv * 16 + (lane & 15);
  f32x8 acc = {0.f, 0.f, 0.f, 0.f, 0.f, 0.f, 0.f, 0.f};
#pragma unroll
  for (int kk = 0; kk < 2; ++kk) {
    bf16x16 a = load_a_frag<true>(arow + kk * 32, lane);
    bf16x16 b = load_b_frag(wbf + colg * HD + kk * 32, lane);
    acc = __builtin_amdgcn_wmma_f32_16x16x32_bf16(false, a, false, b, (short)0, acc, false, false);
  }
  int rbase = m0 + ((lane & 16) ? 8 : 0);
#pragma unroll
  for (int v = 0; v < 8; ++v) hout[(size_t)(rbase + v) * HD + colg] = acc[v];
}
// seq[:,t,:] = layer_norm(leaky_relu(hin)) ; 4 nodes x 64 feats per block
__global__ void k_leaky_ln(const float* __restrict__ hin, const float* __restrict__ g,
                           const float* __restrict__ b, float* __restrict__ seq, int t) {
  __shared__ float s1[256], s2[256];
  int node = blockIdx.x * 4 + (threadIdx.x >> 6);
  int f = threadIdx.x & 63;
  float v = 0.f;
  if (node < N_NODES) v = leaky01(hin[(size_t)node * HD + f]);
  s1[threadIdx.x] = v;
  s2[threadIdx.x] = v * v;
  __syncthreads();
#pragma unroll
  for (int st = 32; st > 0; st >>= 1) {
    if (f < st) {
      s1[threadIdx.x] += s1[threadIdx.x + st];
      s2[threadIdx.x] += s2[threadIdx.x + st];
    }
    __syncthreads();
  }
  int base = threadIdx.x & ~63;
  float mu = s1[base] * (1.f / HD);
  float var = s2[base] * (1.f / HD) - mu * mu;
  if (node < N_NODES) {
    float o = (v - mu) * rsqrtf(var + LN_EPS) * g[f] + b[f];
    seq[(size_t)node * (T_STEPS * HD) + t * HD + f] = o;
  }
}

// ---------------- fused LSTM step ----------------
// gates = [x_t | h_{t-1}] @ [Wih^T ; Whh^T] + bih + bhh ; cell update fused.
// Block = 16 rows x 256 gate cols, 4 waves (wave w -> cols 64w..64w+63).
// h_{t-1} is read from seq[:,t-1,:] (written in-place by the previous step).
__global__ void __launch_bounds__(128) k_lstm_step(float* __restrict__ seq, float* __restrict__ c,
                                                   const unsigned short* __restrict__ wih,
                                                   const unsigned short* __restrict__ whh,
                                                   const float* __restrict__ bih,
                                                   const float* __restrict__ bhh, int t) {
  const int S = 260;  // LDS row stride (pad to dodge bank conflicts)
  __shared__ float lds[16 * S];
  int lane = threadIdx.x & 31, wv = threadIdx.x >> 5;
  int m0 = blockIdx.x * 16;
  const float* xrow = seq + (size_t)(m0 + (lane & 15)) * (T_STEPS * HD) + t * HD;
  f32x8 acc[4];
#pragma unroll
  for (int ct = 0; ct < 4; ++ct) acc[ct] = (f32x8){0.f, 0.f, 0.f, 0.f, 0.f, 0.f, 0.f, 0.f};

#pragma unroll
  for (int kk = 0; kk < 2; ++kk) {  // K rows 0..63: x_t @ Wih^T
    bf16x16 a = load_a_frag<false>(xrow + kk * 32, lane);
#pragma unroll
    for (int ct = 0; ct < 4; ++ct) {
      int colg = wv * 64 + ct * 16 + (lane & 15);
      bf16x16 b = load_b_frag(wih + colg * HD + kk * 32, lane);
      acc[ct] = __builtin_amdgcn_wmma_f32_16x16x32_bf16(false, a, false, b, (short)0, acc[ct], false, false);
    }
  }
  if (t > 0) {  // K rows 64..127: h_{t-1} @ Whh^T (h==0 at t==0, skip)
    const float* hrow = xrow - HD;  // seq[:, t-1, :]
#pragma unroll
    for (int kk = 0; kk < 2; ++kk) {
      bf16x16 a = load_a_frag<false>(hrow + kk * 32, lane);
#pragma unroll
      for (int ct = 0; ct < 4; ++ct) {
        int colg = wv * 64 + ct * 16 + (lane & 15);
        bf16x16 b = load_b_frag(whh + colg * HD + kk * 32, lane);
        acc[ct] = __builtin_amdgcn_wmma_f32_16x16x32_bf16(false, a, false, b, (short)0, acc[ct], false, false);
      }
    }
  }

  // Regroup i/f/g/o columns across waves via LDS.
  int rb = (lane & 16) ? 8 : 0;
#pragma unroll
  for (int ct = 0; ct < 4; ++ct) {
    int col = wv * 64 + ct * 16 + (lane & 15);
#pragma unroll
    for (int v = 0; v < 8; ++v) lds[(rb + v) * S + col] = acc[ct][v];
  }
  __syncthreads();

#pragma unroll
  for (int i = 0; i < 8; ++i) {
    int idx = threadIdx.x + i * 128;  // 16*64 elements / 128 threads
    int row = idx >> 6, j = idx & 63;
    float gi = lds[row * S + j]        + bih[j]        + bhh[j];
    float gf = lds[row * S + 64 + j]   + bih[64 + j]   + bhh[64 + j];
    float gg = lds[row * S + 128 + j]  + bih[128 + j]  + bhh[128 + j];
    float go = lds[row * S + 192 + j]  + bih[192 + j]  + bhh[192 + j];
    size_t n = (size_t)(m0 + row);
    float c_old = (t == 0) ? 0.f : c[n * HD + j];
    float cn = sigm(gf) * c_old + sigm(gi) * tanhf(gg);
    float hn = sigm(go) * tanhf(cn);
    c[n * HD + j] = cn;
    seq[n * (T_STEPS * HD) + t * HD + j] = hn;  // h_t (also next step's h_prev)
  }
}

// ---------------- output heads ----------------
__global__ void k_heads(const float* __restrict__ seq,
                        const float* __restrict__ Wm1, const float* __restrict__ bm1,
                        const float* __restrict__ Wm2, const float* __restrict__ bm2,
                        const float* __restrict__ Wd1, const float* __restrict__ bd1,
                        const float* __restrict__ Wd2, const float* __restrict__ bd2,
                        float* __restrict__ out) {
  int n = blockIdx.x * 256 + threadIdx.x;
  if (n >= N_NODES) return;
  const float* fp = seq + (size_t)n * (T_STEPS * HD) + (T_STEPS - 1) * HD;
  float fr[HD];
#pragma unroll
  for (int k = 0; k < HD; ++k) fr[k] = fp[k];

  float m1[32];
#pragma unroll
  for (int j = 0; j < 32; ++j) {
    float s = bm1[j];
#pragma unroll
    for (int k = 0; k < HD; ++k) s += fr[k] * Wm1[k * 32 + j];
    m1[j] = leaky01(s);
  }
#pragma unroll
  for (int j = 0; j < 4; ++j) {
    float s = bm2[j];
#pragma unroll
    for (int k = 0; k < 32; ++k) s += m1[k] * Wm2[k * 4 + j];
    out[(size_t)n * 4 + j] = fmaxf(s, 0.f) + 1e-4f;
  }
  float d1[16];
#pragma unroll
  for (int j = 0; j < 16; ++j) {
    float s = bd1[j];
#pragma unroll
    for (int k = 0; k < HD; ++k) s += fr[k] * Wd1[k * 16 + j];
    d1[j] = fmaxf(s, 0.f);
  }
#pragma unroll
  for (int j = 0; j < 4; ++j) {
    float s = bd2[j];
#pragma unroll
    for (int k = 0; k < 16; ++k) s += d1[k] * Wd2[k * 4 + j];
    out[(size_t)N_NODES * 4 + (size_t)n * 4 + j] = s;
  }
}

extern "C" void kernel_launch(void* const* d_in, const int* in_sizes, int n_in,
                              void* d_out, int out_size, void* d_ws, size_t ws_size,
                              hipStream_t stream) {
  (void)in_sizes; (void)n_in; (void)out_size; (void)ws_size;
  const float* x    = (const float*)d_in[0];
  const int*   ei   = (const int*)d_in[1];
  const float* ew   = (const float*)d_in[2];
  const float* Wg1  = (const float*)d_in[3];
  const float* bg1  = (const float*)d_in[4];
  const float* Wg2  = (const float*)d_in[5];
  const float* bg2  = (const float*)d_in[6];
  const float* lng  = (const float*)d_in[7];
  const float* lnb  = (const float*)d_in[8];
  const float* Wih0 = (const float*)d_in[9];
  const float* Whh0 = (const float*)d_in[10];
  const float* bih0 = (const float*)d_in[11];
  const float* bhh0 = (const float*)d_in[12];
  const float* Wih1 = (const float*)d_in[13];
  const float* Whh1 = (const float*)d_in[14];
  const float* bih1 = (const float*)d_in[15];
  const float* bhh1 = (const float*)d_in[16];
  const float* Wm1  = (const float*)d_in[17];
  const float* bm1  = (const float*)d_in[18];
  const float* Wm2  = (const float*)d_in[19];
  const float* bm2  = (const float*)d_in[20];
  const float* Wd1  = (const float*)d_in[21];
  const float* bd1  = (const float*)d_in[22];
  const float* Wd2  = (const float*)d_in[23];
  const float* bd2  = (const float*)d_in[24];
  float* out = (float*)d_out;

  char* ws = (char*)d_ws;
  size_t off = 0;
  auto alloc = [&](size_t bytes) -> void* {
    void* p = ws + off;
    off = (off + bytes + 255) & ~(size_t)255;
    return p;
  };
  float* deg  = (float*)alloc((size_t)N_NODES * 4);
  float* norm = (float*)alloc((size_t)(N_EDGES + N_NODES) * 4);
  float* h0   = (float*)alloc((size_t)N_NODES * HD * 4);
  float* h1   = (float*)alloc((size_t)N_NODES * HD * 4);
  float* seq  = (float*)alloc((size_t)N_NODES * T_STEPS * HD * 4);
  unsigned short* wg2bf  = (unsigned short*)alloc((size_t)HD * HD * 2);
  unsigned short* wih0bf = (unsigned short*)alloc((size_t)4 * HD * HD * 2);
  unsigned short* whh0bf = (unsigned short*)alloc((size_t)4 * HD * HD * 2);
  unsigned short* wih1bf = (unsigned short*)alloc((size_t)4 * HD * HD * 2);
  unsigned short* whh1bf = (unsigned short*)alloc((size_t)4 * HD * HD * 2);
  float* cst = h0;  // LSTM cell state reuses h0 (free after GCN phase)

  const int WN = 4 * HD * HD;
  k_f32_to_bf16<<<(WN + 255) / 256, 256, 0, stream>>>(Wih0, wih0bf, WN);
  k_f32_to_bf16<<<(WN + 255) / 256, 256, 0, stream>>>(Whh0, whh0bf, WN);
  k_f32_to_bf16<<<(WN + 255) / 256, 256, 0, stream>>>(Wih1, wih1bf, WN);
  k_f32_to_bf16<<<(WN + 255) / 256, 256, 0, stream>>>(Whh1, whh1bf, WN);
  k_wg2_transpose_bf16<<<(HD * HD + 255) / 256, 256, 0, stream>>>(Wg2, wg2bf);

  k_deg_init<<<(N_NODES + 255) / 256, 256, 0, stream>>>(deg);
  k_deg_edge<<<(N_EDGES + 255) / 256, 256, 0, stream>>>(ei, ew, deg);
  k_dinv_selfnorm<<<(N_NODES + 255) / 256, 256, 0, stream>>>(deg, norm);
  k_norm_edge<<<(N_EDGES + 255) / 256, 256, 0, stream>>>(ei, ew, deg, norm);

  const int nhBlocks = (N_NODES * HD + 255) / 256;
  const int scBlocks = (N_EDGES + N_NODES + 3) / 4;
  const int mBlocks  = N_NODES / 16;  // 50000 % 16 == 0
  for (int t = 0; t < T_STEPS; ++t) {
    k_xw1<<<nhBlocks, 256, 0, stream>>>(x, Wg1, h0, t);
    k_init_bias<<<nhBlocks, 256, 0, stream>>>(h1, bg1);
    k_scatter<<<scBlocks, 256, 0, stream>>>(ei, norm, h0, h1);
    k_gemm_gcn<<<mBlocks, 128, 0, stream>>>(h1, wg2bf, h0);
    k_init_bias<<<nhBlocks, 256, 0, stream>>>(h1, bg2);
    k_scatter<<<scBlocks, 256, 0, stream>>>(ei, norm, h0, h1);
    k_leaky_ln<<<(N_NODES + 3) / 4, 256, 0, stream>>>(h1, lng, lnb, seq, t);
  }

  for (int t = 0; t < T_STEPS; ++t)
    k_lstm_step<<<mBlocks, 128, 0, stream>>>(seq, cst, wih0bf, whh0bf, bih0, bhh0, t);
  for (int t = 0; t < T_STEPS; ++t)
    k_lstm_step<<<mBlocks, 128, 0, stream>>>(seq, cst, wih1bf, whh1bf, bih1, bhh1, t);

  k_heads<<<(N_NODES + 255) / 256, 256, 0, stream>>>(seq, Wm1, bm1, Wm2, bm2,
                                                     Wd1, bd1, Wd2, bd2, out);
}
